// StackedMinGRULM_60318520705470
// MI455X (gfx1250) — matrix operand order repaired and checked by reading, not compiled, not run on hardware
//
#include <hip/hip_runtime.h>

#define DEV __device__ __forceinline__

typedef __attribute__((ext_vector_type(16))) __bf16 v16bf;
typedef __attribute__((ext_vector_type(8)))  __bf16 v8bf;
typedef __attribute__((ext_vector_type(8)))  float  v8f;

union ABFrag { v16bf v; v8bf h[2]; };
union Pk4 { __bf16 e[4]; uint2 u; };

static constexpr int Bb = 8, Sq = 4096, Dm = 512, Ly = 4;
static constexpr int DI = 768, E2 = 1536, FFH = 2048;
static constexpr int Mrows = Bb * Sq;      // 32768
static constexpr int CH = 32;              // scan chunks per sequence
static constexpr int CT = Sq / CH;         // 128 steps per chunk

DEV float softplusf(float x) {
    if (x > 20.f)  return x;
    if (x < -20.f) return expf(x);
    return log1pf(expf(x));
}
DEV float log_g(float x) {
    // log(where(x>=0, x+0.5, sigmoid(x)))
    return (x >= 0.f) ? logf(x + 0.5f) : -softplusf(-x);
}
DEV float logaddexpf(float a, float b) {
    float mx = fmaxf(a, b);
    return mx + logf(expf(a - mx) + expf(b - mx));
}
DEV float gelu_exact(float x) {
    return 0.5f * x * (1.0f + erff(x * 0.70710678118654752f));
}

// ---------------------------------------------------------------------------
// fp32 -> bf16 conversion (weights, once per launch)
// ---------------------------------------------------------------------------
__global__ void cvt_f32_to_bf16(const float* __restrict__ in,
                                __bf16* __restrict__ out, size_t n) {
    size_t i  = (size_t)blockIdx.x * blockDim.x + threadIdx.x;
    size_t st = (size_t)gridDim.x * blockDim.x;
    for (; i < n; i += st) out[i] = (__bf16)in[i];
}

// ---------------------------------------------------------------------------
// LayerNorm over D=512, one wave32 per row, bf16 output (GEMM A operand)
// ---------------------------------------------------------------------------
__global__ __launch_bounds__(256)
void ln512_to_bf16(const float* __restrict__ x, const float* __restrict__ g,
                   const float* __restrict__ b, __bf16* __restrict__ out) {
    const int row  = blockIdx.x * 8 + (threadIdx.x >> 5);
    const int lane = threadIdx.x & 31;
    const float4* xr = (const float4*)(x + (size_t)row * Dm);
    float4 v[4];
    float s = 0.f, q = 0.f;
#pragma unroll
    for (int k = 0; k < 4; k++) {
        v[k] = xr[lane + 32 * k];
        s += v[k].x + v[k].y + v[k].z + v[k].w;
        q += v[k].x * v[k].x + v[k].y * v[k].y + v[k].z * v[k].z + v[k].w * v[k].w;
    }
#pragma unroll
    for (int off = 16; off > 0; off >>= 1) {
        s += __shfl_xor(s, off);
        q += __shfl_xor(q, off);
    }
    const float m  = s * (1.f / Dm);
    const float vr = q * (1.f / Dm) - m * m;
    const float rs = rsqrtf(vr + 1e-5f);
    uint2* orow = (uint2*)(out + (size_t)row * Dm);
#pragma unroll
    for (int k = 0; k < 4; k++) {
        float4 gv = ((const float4*)g)[lane + 32 * k];
        float4 bv = ((const float4*)b)[lane + 32 * k];
        Pk4 p;
        p.e[0] = (__bf16)((v[k].x - m) * rs * gv.x + bv.x);
        p.e[1] = (__bf16)((v[k].y - m) * rs * gv.y + bv.y);
        p.e[2] = (__bf16)((v[k].z - m) * rs * gv.z + bv.z);
        p.e[3] = (__bf16)((v[k].w - m) * rs * gv.w + bv.w);
        orow[lane + 32 * k] = p.u;
    }
}

// ---------------------------------------------------------------------------
// WMMA bf16 GEMM:  C[M,N] = A[M,K] @ W[N,K]^T  (+ epilogue)
//   EPI 0: fp32 out (hg)
//   EPI 1: + residual, fp32 out (Wout -> y)
//   EPI 2: + bias, exact GELU, bf16 out (FF1)
//   EPI 3: + bias + residual, fp32 out (FF2 -> x)
// Block tile 128x128, BK=32, 8 waves (4 along M x 2 along N), wave = 32x64.
// Triple-buffered LDS, 2 tiles in flight via GLOBAL_LOAD_ASYNC_TO_LDS_B128
// (ASYNCcnt-tracked, VGPR-bypassing). Each tile = 4 async-load instructions;
// async loads retire in order, so with tiles {k+1,k+2} outstanding,
// s_wait_asynccnt 4 guarantees tile k+1 has landed. The two tail tiles are
// peeled so the steady-state loop body is branch-free (issue / compute /
// wait 4 / barrier). Requires K >= 64, K % 32 == 0 (true for all call sites).
// ---------------------------------------------------------------------------
template <int EPI>
__global__ __launch_bounds__(256)
void gemm_bf16_wmma(const __bf16* __restrict__ A, const __bf16* __restrict__ W,
                    const float* __restrict__ bias, const float* __restrict__ res,
                    float* __restrict__ Cf, __bf16* __restrict__ Cb,
                    int M, int N, int K) {
    __shared__ alignas(128) __bf16 As[3][128 * 32];
    __shared__ alignas(128) __bf16 Ws[3][128 * 32];

    const int tid  = threadIdx.x;
    const int lane = tid & 31;
    const int wid  = tid >> 5;
    const int bm   = blockIdx.y * 128;
    const int bn   = blockIdx.x * 128;
    const int wm   = (wid & 3) * 32;   // wave offset along M
    const int wn   = (wid >> 2) * 64;  // wave offset along N
    const int lrow = tid >> 2;         // 0..63: tile row this thread loads
    const int lcol = (tid & 3) * 8;    // half-element column offset
    const int nm   = lane & 15;        // M (for A) or N (for B/C) within tile
    const int hl   = lane >> 4;        // half-wave select

    const __bf16* gA0 = A + (size_t)(bm + lrow) * K + lcol;
    const __bf16* gA1 = A + (size_t)(bm + 64 + lrow) * K + lcol;
    const __bf16* gW0 = W + (size_t)(bn + lrow) * K + lcol;
    const __bf16* gW1 = W + (size_t)(bn + 64 + lrow) * K + lcol;

    // Flat LDS addresses truncate to the wave-relative LDS byte offset
    // (ISA 10.2: the low 32 address bits).
    auto issue_tile = [&](int k0, int buf) {
        uint32_t dA0 = (uint32_t)(uintptr_t)&As[buf][lrow * 32 + lcol];
        uint32_t dA1 = (uint32_t)(uintptr_t)&As[buf][(64 + lrow) * 32 + lcol];
        uint32_t dW0 = (uint32_t)(uintptr_t)&Ws[buf][lrow * 32 + lcol];
        uint32_t dW1 = (uint32_t)(uintptr_t)&Ws[buf][(64 + lrow) * 32 + lcol];
        asm volatile("global_load_async_to_lds_b128 %0, %1, off"
                     :: "v"(dA0), "v"((uint64_t)(uintptr_t)(gA0 + k0)) : "memory");
        asm volatile("global_load_async_to_lds_b128 %0, %1, off"
                     :: "v"(dA1), "v"((uint64_t)(uintptr_t)(gA1 + k0)) : "memory");
        asm volatile("global_load_async_to_lds_b128 %0, %1, off"
                     :: "v"(dW0), "v"((uint64_t)(uintptr_t)(gW0 + k0)) : "memory");
        asm volatile("global_load_async_to_lds_b128 %0, %1, off"
                     :: "v"(dW1), "v"((uint64_t)(uintptr_t)(gW1 + k0)) : "memory");
    };

    v8f acc[2][4] = {};

    // One K-step of WMMAs from LDS buffer `buf`.
    auto compute_tile = [&](int buf) {
        // A fragments (16x32 bf16): lane m=lane%16, K-half h=lane/16;
        // elements e=0..15 -> K = 16h + e
        ABFrag a[2];
#pragma unroll
        for (int mi = 0; mi < 2; mi++) {
            const __bf16* pr = &As[buf][(wm + mi * 16 + nm) * 32 + 16 * hl];
            a[mi].h[0] = *(const v8bf*)pr;
            a[mi].h[1] = *(const v8bf*)(pr + 8);
        }
        // B fragments (32x16 bf16): lane n=lane%16, K-half h=lane/16;
        // elements e=0..15 -> K = 16h + e (contiguous in W's K-major rows)
#pragma unroll
        for (int ni = 0; ni < 4; ni++) {
            v16bf bfrag = *(const v16bf*)&Ws[buf][(wn + ni * 16 + nm) * 32 + 16 * hl];
#pragma unroll
            for (int mi = 0; mi < 2; mi++) {
                acc[mi][ni] = __builtin_amdgcn_wmma_f32_16x16x32_bf16(
                    false, a[mi].v, false, bfrag, (short)0, acc[mi][ni],
                    false, false);
            }
        }
    };

    // Prologue: put tiles 0 and 1 in flight, wait until tile 0 has landed.
    issue_tile(0, 0);
    issue_tile(32, 1);
    asm volatile("s_wait_asynccnt 4" ::: "memory");
    __syncthreads();

    // Steady state: branch-free issue/compute/wait/barrier per K-step.
    int buf = 0;
    for (int k0 = 0; k0 + 64 < K; k0 += 32) {
        issue_tile(k0 + 64, (buf + 2) % 3);
        compute_tile(buf);
        asm volatile("s_wait_asynccnt 4" ::: "memory");
        __syncthreads();
        buf = (buf + 1) % 3;
    }
    // Tail: two tiles already resident / in flight.
    compute_tile(buf);
    asm volatile("s_wait_asynccnt 0" ::: "memory");
    __syncthreads();
    compute_tile((buf + 1) % 3);

    // Epilogue. C/D layout: lane n=lane%16, h=lane/16; VGPR r -> M = r + 8h
#pragma unroll
    for (int mi = 0; mi < 2; mi++) {
#pragma unroll
        for (int ni = 0; ni < 4; ni++) {
#pragma unroll
            for (int r = 0; r < 8; r++) {
                const int gm = bm + wm + mi * 16 + r + 8 * hl;
                const int gn = bn + wn + ni * 16 + nm;
                float c = acc[mi][ni][r];
                if (EPI == 2 || EPI == 3) c += bias[gn];
                if (EPI == 1 || EPI == 3) c += res[(size_t)gm * N + gn];
                if (EPI == 2) {
                    Cb[(size_t)gm * N + gn] = (__bf16)gelu_exact(c);
                } else {
                    Cf[(size_t)gm * N + gn] = c;
                }
            }
        }
    }
}

// ---------------------------------------------------------------------------
// Chunked Heinsen scan (3 passes). hg[M,1536] holds raw {hidden | gate};
// log-space transforms are computed on the fly (no separate prep pass).
//   lc = -softplus(gate), lv = -softplus(-gate) + log_g(hidden)
// Pass 1: per chunk c of channel (b,e): A_c = sum lc ; L_c = logsumexp(lv - a_loc)
// Pass 2: per channel: P_{c+1}=P_c+A_c ; S_{c+1}=logaddexp(S_c, L_c - P_c),
//         seeded P_0=0, S_0=log(0.5) (the prepended h0 term); emits K_c=P_c+S_c.
// Pass 3: per chunk: stream a_loc/acc (acc seeded K_c), h = exp(a_loc + acc).
// ---------------------------------------------------------------------------
__global__ __launch_bounds__(256)
void scan_pass1(const float* __restrict__ hg, float* __restrict__ Aout,
                float* __restrict__ Lout) {
    const int idx = blockIdx.x * 256 + threadIdx.x;   // (b*CH + c)*DI + e
    if (idx >= Bb * CH * DI) return;
    const int e = idx % DI;
    const int c = (idx / DI) % CH;
    const int b = idx / (DI * CH);
    const float* base = hg + ((size_t)b * Sq + (size_t)c * CT) * E2;
    float a = 0.f, acc = -__builtin_inff();
    for (int t = 0; t < CT; t++) {
        const float* pt = base + (size_t)t * E2;
        const float lc = -softplusf(pt[DI + e]);
        const float lv = -softplusf(-pt[DI + e]) + log_g(pt[e]);
        a += lc;
        acc = logaddexpf(acc, lv - a);
    }
    Aout[idx] = a;
    Lout[idx] = acc;
}

__global__ void scan_pass2(const float* __restrict__ Ain,
                           const float* __restrict__ Lin,
                           float* __restrict__ Kout) {
    const int idx = blockIdx.x * blockDim.x + threadIdx.x;  // b*DI + e
    if (idx >= Bb * DI) return;
    const int e = idx % DI, b = idx / DI;
    float P = 0.f, S = -0.69314718055994531f;   // log 0.5 (prepended h0 term)
    for (int c = 0; c < CH; c++) {
        const size_t o = ((size_t)b * CH + c) * DI + e;
        Kout[o] = P + S;
        S = logaddexpf(S, Lin[o] - P);
        P += Ain[o];
    }
}

__global__ __launch_bounds__(256)
void scan_pass3(const float* __restrict__ hg, const float* __restrict__ Kin,
                __bf16* __restrict__ h) {
    const int idx = blockIdx.x * 256 + threadIdx.x;   // (b*CH + c)*DI + e
    if (idx >= Bb * CH * DI) return;
    const int e = idx % DI;
    const int c = (idx / DI) % CH;
    const int b = idx / (DI * CH);
    const float* base = hg + ((size_t)b * Sq + (size_t)c * CT) * E2;
    __bf16* ho = h + ((size_t)b * Sq + (size_t)c * CT) * DI + e;
    float a = 0.f, acc = Kin[idx];
    for (int t = 0; t < CT; t++) {
        const float* pt = base + (size_t)t * E2;
        const float lc = -softplusf(pt[DI + e]);
        const float lv = -softplusf(-pt[DI + e]) + log_g(pt[e]);
        a += lc;
        acc = logaddexpf(acc, lv - a);
        ho[(size_t)t * DI] = (__bf16)expf(a + acc);
    }
}

// ---------------------------------------------------------------------------
// Final LayerNorm + logits (N=2): wave per row, fused dot products
// ---------------------------------------------------------------------------
__global__ __launch_bounds__(256)
void logits512(const float* __restrict__ x, const float* __restrict__ g,
               const float* __restrict__ b, const float* __restrict__ Wl,
               float* __restrict__ out) {
    const int row  = blockIdx.x * 8 + (threadIdx.x >> 5);
    const int lane = threadIdx.x & 31;
    const float4* xr = (const float4*)(x + (size_t)row * Dm);
    float4 v[4];
    float s = 0.f, q = 0.f;
#pragma unroll
    for (int k = 0; k < 4; k++) {
        v[k] = xr[lane + 32 * k];
        s += v[k].x + v[k].y + v[k].z + v[k].w;
        q += v[k].x * v[k].x + v[k].y * v[k].y + v[k].z * v[k].z + v[k].w * v[k].w;
    }
#pragma unroll
    for (int off = 16; off > 0; off >>= 1) {
        s += __shfl_xor(s, off);
        q += __shfl_xor(q, off);
    }
    const float m  = s * (1.f / Dm);
    const float vr = q * (1.f / Dm) - m * m;
    const float rs = rsqrtf(vr + 1e-5f);
    float d0 = 0.f, d1 = 0.f;
#pragma unroll
    for (int k = 0; k < 4; k++) {
        float4 gv = ((const float4*)g)[lane + 32 * k];
        float4 bv = ((const float4*)b)[lane + 32 * k];
        float4 w0 = ((const float4*)Wl)[lane + 32 * k];
        float4 w1 = ((const float4*)(Wl + Dm))[lane + 32 * k];
        float x0 = (v[k].x - m) * rs * gv.x + bv.x;
        float x1 = (v[k].y - m) * rs * gv.y + bv.y;
        float x2 = (v[k].z - m) * rs * gv.z + bv.z;
        float x3 = (v[k].w - m) * rs * gv.w + bv.w;
        d0 += x0 * w0.x + x1 * w0.y + x2 * w0.z + x3 * w0.w;
        d1 += x0 * w1.x + x1 * w1.y + x2 * w1.z + x3 * w1.w;
    }
#pragma unroll
    for (int off = 16; off > 0; off >>= 1) {
        d0 += __shfl_xor(d0, off);
        d1 += __shfl_xor(d1, off);
    }
    if (lane == 0) {
        out[(size_t)row * 2 + 0] = d0;
        out[(size_t)row * 2 + 1] = d1;
    }
}

// ---------------------------------------------------------------------------
extern "C" void kernel_launch(void* const* d_in, const int* in_sizes, int n_in,
                              void* d_out, int out_size, void* d_ws, size_t ws_size,
                              hipStream_t stream) {
    const float* input = (const float*)d_in[0];
    const float* Whg   = (const float*)d_in[1];
    const float* Wout  = (const float*)d_in[2];
    const float* ln1g  = (const float*)d_in[3];
    const float* ln1b  = (const float*)d_in[4];
    const float* ln2g  = (const float*)d_in[5];
    const float* ln2b  = (const float*)d_in[6];
    const float* ffW1  = (const float*)d_in[7];
    const float* ffb1  = (const float*)d_in[8];
    const float* ffW2  = (const float*)d_in[9];
    const float* ffb2  = (const float*)d_in[10];
    const float* normg = (const float*)d_in[11];
    const float* normb = (const float*)d_in[12];
    const float* Wlog  = (const float*)d_in[13];
    float* out = (float*)d_out;

    // Workspace partition (256B-aligned slabs)
    char* p = (char*)d_ws;
    auto alloc = [&](size_t bytes) -> char* {
        char* r = p;
        p += (bytes + 255) & ~(size_t)255;
        return r;
    };
    const size_t nWhg = (size_t)Ly * E2 * Dm;
    const size_t nWo  = (size_t)Ly * Dm * DI;
    const size_t nW1  = (size_t)Ly * FFH * Dm;
    const size_t nW2  = (size_t)Ly * Dm * FFH;
    __bf16* wWhg = (__bf16*)alloc(nWhg * 2);
    __bf16* wWo  = (__bf16*)alloc(nWo * 2);
    __bf16* wW1  = (__bf16*)alloc(nW1 * 2);
    __bf16* wW2  = (__bf16*)alloc(nW2 * 2);
    float*  xbuf = (float*)alloc((size_t)Mrows * Dm * 4);
    float*  ybuf = (float*)alloc((size_t)Mrows * Dm * 4);
    float*  hgbf = (float*)alloc((size_t)Mrows * E2 * 4);
    __bf16* nbuf = (__bf16*)alloc((size_t)Mrows * Dm * 2);    // LN outputs
    __bf16* hbf  = (__bf16*)alloc((size_t)Mrows * DI * 2);    // scan output
    __bf16* abuf = (__bf16*)alloc((size_t)Mrows * FFH * 2);   // gelu output
    float*  scA  = (float*)alloc((size_t)Bb * CH * DI * 4);   // chunk lc-sums
    float*  scL  = (float*)alloc((size_t)Bb * CH * DI * 4);   // chunk logsumexp
    float*  scK  = (float*)alloc((size_t)Bb * CH * DI * 4);   // chunk carries

    // Weights -> bf16 (per call; deterministic)
    cvt_f32_to_bf16<<<1024, 256, 0, stream>>>(Whg, wWhg, nWhg);
    cvt_f32_to_bf16<<<1024, 256, 0, stream>>>(Wout, wWo, nWo);
    cvt_f32_to_bf16<<<1024, 256, 0, stream>>>(ffW1, wW1, nW1);
    cvt_f32_to_bf16<<<1024, 256, 0, stream>>>(ffW2, wW2, nW2);

    hipMemcpyAsync(xbuf, input, (size_t)Mrows * Dm * sizeof(float),
                   hipMemcpyDeviceToDevice, stream);

    const int scanGrid = (Bb * CH * DI + 255) / 256;   // 768 blocks
    for (int i = 0; i < Ly; i++) {
        // LN1 -> bf16
        ln512_to_bf16<<<Mrows / 8, 256, 0, stream>>>(
            xbuf, ln1g + (size_t)i * Dm, ln1b + (size_t)i * Dm, nbuf);
        // hg = LN1(x) @ Whg^T    [M,1536] fp32 (raw hidden|gate)
        gemm_bf16_wmma<0><<<dim3(E2 / 128, Mrows / 128), 256, 0, stream>>>(
            nbuf, wWhg + (size_t)i * E2 * Dm, nullptr, nullptr,
            hgbf, nullptr, Mrows, E2, Dm);
        // chunked log-space scan -> h bf16 [M,768]
        scan_pass1<<<scanGrid, 256, 0, stream>>>(hgbf, scA, scL);
        scan_pass2<<<(Bb * DI + 255) / 256, 256, 0, stream>>>(scA, scL, scK);
        scan_pass3<<<scanGrid, 256, 0, stream>>>(hgbf, scK, hbf);
        // y = h @ Wout^T + x     [M,512] fp32
        gemm_bf16_wmma<1><<<dim3(Dm / 128, Mrows / 128), 256, 0, stream>>>(
            hbf, wWo + (size_t)i * Dm * DI, nullptr, xbuf,
            ybuf, nullptr, Mrows, Dm, DI);
        // LN2 -> bf16
        ln512_to_bf16<<<Mrows / 8, 256, 0, stream>>>(
            ybuf, ln2g + (size_t)i * Dm, ln2b + (size_t)i * Dm, nbuf);
        // ff = gelu(LN2(y) @ W1^T + b1)   bf16 [M,2048]
        gemm_bf16_wmma<2><<<dim3(FFH / 128, Mrows / 128), 256, 0, stream>>>(
            nbuf, wW1 + (size_t)i * FFH * Dm, ffb1 + (size_t)i * FFH, nullptr,
            nullptr, abuf, Mrows, FFH, Dm);
        // x = ff @ W2^T + b2 + y          fp32 [M,512]
        gemm_bf16_wmma<3><<<dim3(Dm / 128, Mrows / 128), 256, 0, stream>>>(
            abuf, wW2 + (size_t)i * Dm * FFH, ffb2 + (size_t)i * Dm, ybuf,
            xbuf, nullptr, Mrows, Dm, FFH);
    }

    // Final LN + logits (N=2)
    logits512<<<Mrows / 8, 256, 0, stream>>>(xbuf, normg, normb, Wlog, out);

    (void)in_sizes; (void)n_in; (void)out_size; (void)ws_size;
}
